// GaT_61478161875327
// MI455X (gfx1250) — matrix operand order repaired and checked by reading, not compiled, not run on hardware
//
#include <hip/hip_runtime.h>

// ---------------------------------------------------------------------------
// CDNA5 (gfx1250, wave32) fused cross-attention + dense-MoE block.
// - All GEMMs on v_wmma_f32_16x16x32_bf16 (f32 accumulate)
// - Fragment loads are 2x ds_load_b128 (A row-major, B K-transposed in LDS)
// - One-shot tile staging (Q tile, out-proj A tile) uses
//   GLOBAL_LOAD_ASYNC_TO_LDS_B128 (ASYNCcnt-tracked, no VGPR round-trip).
//   In-loop staging is synchronous and NOT unrolled: a compact rolled loop
//   keeps only one address/data set live so the softmax state stays in
//   registers (no per-chunk spills).
// - MoE folds the per-token gate into the hidden activations so the second
//   GEMM accumulates straight into the output accumulator (no yacc).
// ---------------------------------------------------------------------------

typedef __attribute__((ext_vector_type(16))) __bf16 v16bf;
typedef __attribute__((ext_vector_type(8)))  __bf16 v8bf;
typedef __attribute__((ext_vector_type(8)))  float  v8f;
typedef __attribute__((ext_vector_type(4)))  int    v4i;

#define AS_GLOBAL __attribute__((address_space(1)))
#define AS_LOCAL  __attribute__((address_space(3)))

#define DD   256      // model dim
#define HH   256      // attention inner dim (single head)
#define EE   8        // experts
#define DFF  1024     // expert hidden dim
#define BB   32
#define LSS  1024
#define LGG  512
#define NEGV (-1e9f)

#if defined(__AMDGCN__) &&                                                    \
    __has_builtin(__builtin_amdgcn_global_load_async_to_lds_b128) &&          \
    __has_builtin(__builtin_amdgcn_s_wait_asynccnt)
#define HAS_ASYNC_LDS 1
#else
#define HAS_ASYNC_LDS 0
#endif

// ---- WMMA helper -----------------------------------------------------------
__device__ __forceinline__ v8f wmma_bf16(v16bf a, v16bf b, v8f c) {
  return __builtin_amdgcn_wmma_f32_16x16x32_bf16(false, a, false, b, (short)0, c,
                                                 false, false);
}

// ---- Fragment loaders: two ds_load_b128 each -------------------------------
__device__ __forceinline__ v16bf load_a_frag(const __bf16* As, int lda, int k0,
                                             int lane) {
  int mrow = lane & 15, half = lane >> 4;
  const __bf16* p = As + mrow * lda + k0 + half * 16;
  v8bf lo = *(const v8bf*)p;
  v8bf hi = *(const v8bf*)(p + 8);
  return __builtin_shufflevector(lo, hi, 0, 1, 2, 3, 4, 5, 6, 7, 8, 9, 10, 11,
                                 12, 13, 14, 15);
}

__device__ __forceinline__ v16bf load_bt_frag(const __bf16* Bt, int ldk, int n0,
                                              int k0, int lane) {
  int ncol = n0 + (lane & 15), half = lane >> 4;
  const __bf16* p = Bt + ncol * ldk + k0 + half * 16;
  v8bf lo = *(const v8bf*)p;
  v8bf hi = *(const v8bf*)(p + 8);
  return __builtin_shufflevector(lo, hi, 0, 1, 2, 3, 4, 5, 6, 7, 8, 9, 10, 11,
                                 12, 13, 14, 15);
}

__device__ __forceinline__ float gelu_tanh(float x) {
  float x3 = x * x * x;
  return 0.5f * x * (1.0f + tanhf(0.7978845608f * (x + 0.044715f * x3)));
}

// ---- staging helpers -------------------------------------------------------
// f32 global -> bf16 LDS, 8 elements (two float4 loads, one b128 store)
__device__ __forceinline__ void stage8_f32(const float* __restrict__ g,
                                           __bf16* l) {
  const float4 f0 = *(const float4*)g;
  const float4 f1 = *(const float4*)(g + 4);
  v8bf v = {(__bf16)f0.x, (__bf16)f0.y, (__bf16)f0.z, (__bf16)f0.w,
            (__bf16)f1.x, (__bf16)f1.y, (__bf16)f1.z, (__bf16)f1.w};
  *(v8bf*)l = v;
}
// contiguous 16B bf16 global -> LDS, synchronous (VGPR round-trip)
__device__ __forceinline__ void stage8_bf16(const __bf16* __restrict__ g,
                                            __bf16* l) {
  *(v8bf*)l = *(const v8bf*)g;
}
// contiguous 16B bf16 global -> LDS, async (no VGPR round-trip). Use only in
// one-shot prologues (before accumulators are live) to avoid spills.
__device__ __forceinline__ void stage8_bf16_async(const __bf16* g, __bf16* l) {
#if HAS_ASYNC_LDS
  __builtin_amdgcn_global_load_async_to_lds_b128((AS_GLOBAL v4i*)g,
                                                 (AS_LOCAL v4i*)l, 0, 0);
#else
  *(v8bf*)l = *(const v8bf*)g;
#endif
}
__device__ __forceinline__ void async_wait() {
#if HAS_ASYNC_LDS
  __builtin_amdgcn_s_wait_asynccnt(0);
#endif
}
// 8 strided global bf16 reads -> one contiguous 16B LDS write (transpose)
__device__ __forceinline__ void stage8_strided(const __bf16* __restrict__ g,
                                               size_t stride, __bf16* l) {
  v8bf v = {g[0], g[stride], g[2 * stride], g[3 * stride], g[4 * stride],
            g[5 * stride], g[6 * stride], g[7 * stride]};
  *(v8bf*)l = v;
}

// ---------------------------------------------------------------------------
// Kernel 0: f32 -> bf16 weight conversion
// ---------------------------------------------------------------------------
__global__ void cvt_f32_bf16_kernel(const float* __restrict__ src,
                                    __bf16* __restrict__ dst, int n) {
  int i = (blockIdx.x * 256 + threadIdx.x) * 8;
  if (i < n) {
    const float4 f0 = *(const float4*)(src + i);
    const float4 f1 = *(const float4*)(src + i + 4);
    v8bf v = {(__bf16)f0.x, (__bf16)f0.y, (__bf16)f0.z, (__bf16)f0.w,
              (__bf16)f1.x, (__bf16)f1.y, (__bf16)f1.z, (__bf16)f1.w};
    *(v8bf*)(dst + i) = v;
  }
}

// ---------------------------------------------------------------------------
// Kernel 1: 256x256-weight GEMM, tile 64(M) x 64(N), K=256.
// ---------------------------------------------------------------------------
template <bool A_BF16, bool RESID>
__global__ void gemm256_kernel(const void* __restrict__ Ap,
                               const __bf16* __restrict__ Bw,
                               const float* __restrict__ resid,
                               void* __restrict__ outp, int M) {
  __shared__ alignas(16) __bf16 As[64][DD];   // 32 KB row-major
  __shared__ alignas(16) __bf16 Bt[64][DD];   // 32 KB [n][k]
  int tid = threadIdx.x, lane = tid & 31, wave = tid >> 5;
  int wm = wave & 3, wn = wave >> 2;
  int m0 = blockIdx.x * 64, n0 = blockIdx.y * 64;
  (void)M;

  if (A_BF16) {
    const __bf16* A = (const __bf16*)Ap + (size_t)m0 * DD;
#pragma unroll
    for (int c = 0; c < 64 * DD / 8 / 256; ++c)
      stage8_bf16_async(A + (tid + c * 256) * 8,
                        &As[0][0] + (tid + c * 256) * 8);
  } else {
    const float* A = (const float*)Ap + (size_t)m0 * DD;
#pragma unroll
    for (int c = 0; c < 64 * DD / 8 / 256; ++c)
      stage8_f32(A + (tid + c * 256) * 8, &As[0][0] + (tid + c * 256) * 8);
  }
#pragma unroll
  for (int cc = 0; cc < 64 * (DD / 8) / 256; ++cc) {
    int c = tid + cc * 256;
    int n = c >> 5, kc = (c & 31) * 8;
    stage8_strided(Bw + (size_t)kc * DD + n0 + n, DD, &Bt[n][kc]);
  }
  async_wait();
  __syncthreads();

  v8f acc[2] = {};
  for (int k0 = 0; k0 < DD; k0 += 32) {
    v16bf a = load_a_frag(&As[wm * 16][0], DD, k0, lane);
#pragma unroll
    for (int nt = 0; nt < 2; ++nt) {
      v16bf b = load_bt_frag(&Bt[0][0], DD, wn * 32 + nt * 16, k0, lane);
      acc[nt] = wmma_bf16(a, b, acc[nt]);
    }
  }

  int half = lane >> 4;
#pragma unroll
  for (int nt = 0; nt < 2; ++nt) {
#pragma unroll
    for (int r = 0; r < 8; ++r) {
      size_t row = (size_t)m0 + wm * 16 + r + 8 * half;
      int col = n0 + wn * 32 + nt * 16 + (lane & 15);
      float v = acc[nt][r];
      if (RESID) v += resid[row * DD + col];
      if (A_BF16)
        ((float*)outp)[row * DD + col] = v;
      else
        ((__bf16*)outp)[row * DD + col] = (__bf16)v;
    }
  }
}

// ---------------------------------------------------------------------------
// Kernel 2: flash-style cross attention. 64-query tile/block, K/V streamed in
// 32-key chunks, online softmax (wave32 shfl). Q staged async (one-shot);
// in-loop staging is a compact rolled loop to minimize live registers.
// ---------------------------------------------------------------------------
__global__ void attn_kernel(const __bf16* __restrict__ Q,
                            const __bf16* __restrict__ K,
                            const __bf16* __restrict__ V,
                            const unsigned char* __restrict__ mask_kv,
                            __bf16* __restrict__ ctx, int nQ, int nKV) {
  __shared__ alignas(16) __bf16 Qs[64][HH];     // 32 KB row-major (A)
  __shared__ alignas(16) __bf16 Ksm[32][HH];    // 16 KB [key][h] (Bt for S)
  __shared__ alignas(16) __bf16 Vt[HH][32];     // 16 KB [h][key] (Bt for P@V)
  __shared__ alignas(16) __bf16 Ps[4][16][32];  //  4 KB row-major (A)

  int b = blockIdx.y;
  int q0 = blockIdx.x * 64;
  const __bf16* Qb = Q + ((size_t)b * nQ + q0) * HH;
  const __bf16* Kb = K + (size_t)b * nKV * HH;
  const __bf16* Vb = V + (size_t)b * nKV * HH;
  const unsigned char* mb = mask_kv + (size_t)b * nKV;

  int tid = threadIdx.x, lane = tid & 31, wave = tid >> 5;
  int wm = wave & 3, wn = wave >> 2, half = lane >> 4;

#pragma unroll
  for (int c = 0; c < 64 * HH / 8 / 256; ++c)
    stage8_bf16_async(Qb + (tid + c * 256) * 8,
                      &Qs[0][0] + (tid + c * 256) * 8);
  async_wait();
  __syncthreads();

  v8f ctxacc[8] = {};
  float rm[8], rl[8];
#pragma unroll
  for (int r = 0; r < 8; ++r) { rm[r] = -3.0e30f; rl[r] = 0.f; }

  const float scale = 0.0625f;  // 1/sqrt(256)

  for (int m0 = 0; m0 < nKV; m0 += 32) {
    // rolled staging loops: one address/data set live at a time
    for (int c = tid; c < 32 * HH / 8; c += 256)
      stage8_bf16(Kb + (size_t)m0 * HH + c * 8, &Ksm[0][0] + c * 8);
    for (int c = tid; c < HH * (32 / 8); c += 256) {
      int h = c >> 2, kc = (c & 3) * 8;
      stage8_strided(Vb + (size_t)(m0 + kc) * HH + h, HH, &Vt[h][kc]);
    }
    __syncthreads();

    // S(16x32) = Q_tile @ K_chunk^T  (both wn waves compute it redundantly)
    v8f s[2] = {};
    for (int k0 = 0; k0 < HH; k0 += 32) {
      v16bf a = load_a_frag(&Qs[wm * 16][0], HH, k0, lane);
#pragma unroll
      for (int nt = 0; nt < 2; ++nt) {
        v16bf bt = load_bt_frag(&Ksm[0][0], HH, nt * 16, k0, lane);
        s[nt] = wmma_bf16(a, bt, s[nt]);
      }
    }
#pragma unroll
    for (int nt = 0; nt < 2; ++nt) {
      bool pad = mb[m0 + nt * 16 + (lane & 15)] != 0;
#pragma unroll
      for (int r = 0; r < 8; ++r) s[nt][r] = pad ? NEGV : s[nt][r] * scale;
    }
    // online softmax; rescale factor applied inline
#pragma unroll
    for (int r = 0; r < 8; ++r) {
      float mx = fmaxf(s[0][r], s[1][r]);
      for (int off = 1; off < 16; off <<= 1) mx = fmaxf(mx, __shfl_xor(mx, off));
      float nm = fmaxf(rm[r], mx);
      s[0][r] = __expf(s[0][r] - nm);
      s[1][r] = __expf(s[1][r] - nm);
      float sum = s[0][r] + s[1][r];
      for (int off = 1; off < 16; off <<= 1) sum += __shfl_xor(sum, off);
      float corr = __expf(rm[r] - nm);
      rl[r] = rl[r] * corr + sum;
      rm[r] = nm;
#pragma unroll
      for (int j = 0; j < 8; ++j) ctxacc[j][r] *= corr;
    }

    if (wn == 0) {
#pragma unroll
      for (int nt = 0; nt < 2; ++nt)
#pragma unroll
        for (int r = 0; r < 8; ++r)
          Ps[wm][r + 8 * half][nt * 16 + (lane & 15)] = (__bf16)s[nt][r];
    }
    __syncthreads();

    {
      v16bf a = load_a_frag(&Ps[wm][0][0], 32, 0, lane);
#pragma unroll
      for (int j = 0; j < 8; ++j) {
        v16bf bv = load_bt_frag(&Vt[0][0], 32, wn * 128 + j * 16, 0, lane);
        ctxacc[j] = wmma_bf16(a, bv, ctxacc[j]);
      }
    }
    __syncthreads();
  }

#pragma unroll
  for (int j = 0; j < 8; ++j) {
#pragma unroll
    for (int r = 0; r < 8; ++r) {
      float inv = 1.0f / rl[r];
      size_t row = (size_t)b * nQ + q0 + wm * 16 + r + 8 * half;
      int col = wn * 128 + j * 16 + (lane & 15);
      ctx[row * HH + col] = (__bf16)(ctxacc[j][r] * inv);
    }
  }
}

// ---------------------------------------------------------------------------
// Kernel 3: gate softmax
// ---------------------------------------------------------------------------
__global__ void gate_kernel(const float* __restrict__ X,
                            const float* __restrict__ Wg,
                            float* __restrict__ gates, int T) {
  int t = blockIdx.x * 256 + threadIdx.x;
  if (t >= T) return;
  const float* x = X + (size_t)t * DD;
  float acc[EE] = {};
  for (int d = 0; d < DD; ++d) {
    float xv = x[d];
#pragma unroll
    for (int e = 0; e < EE; ++e) acc[e] += xv * Wg[d * EE + e];
  }
  float mx = acc[0];
#pragma unroll
  for (int e = 1; e < EE; ++e) mx = fmaxf(mx, acc[e]);
  float sum = 0.f;
#pragma unroll
  for (int e = 0; e < EE; ++e) { acc[e] = __expf(acc[e] - mx); sum += acc[e]; }
  float inv = 1.0f / sum;
#pragma unroll
  for (int e = 0; e < EE; ++e) gates[(size_t)t * EE + e] = acc[e] * inv;
}

// ---------------------------------------------------------------------------
// Kernel 4: fused dense-MoE, gate folded into hidden activations.
// ---------------------------------------------------------------------------
__global__ void moe_kernel(const float* __restrict__ X,
                           const __bf16* __restrict__ W1,
                           const float* __restrict__ B1,
                           const __bf16* __restrict__ W2,
                           const float* __restrict__ B2,
                           const float* __restrict__ gates,
                           float* __restrict__ out, int T) {
  __shared__ alignas(16) __bf16 Xs[64][DD];    // 32 KB row-major (A)
  __shared__ alignas(16) __bf16 W1t[64][DD];   // 32 KB [n=f][k=d]
  __shared__ alignas(16) __bf16 W2t[DD][64];   // 32 KB [n=d][k=f]
  __shared__ alignas(16) __bf16 Hs[64][64];    //  8 KB row-major (A)
  (void)T;

  int tid = threadIdx.x, lane = tid & 31, wave = tid >> 5;
  int wm = wave & 3, wn = wave >> 2, half = lane >> 4;
  int t0 = blockIdx.x * 64;

  {
    const float* Xb = X + (size_t)t0 * DD;
#pragma unroll
    for (int c = 0; c < 64 * DD / 8 / 256; ++c)
      stage8_f32(Xb + (tid + c * 256) * 8, &Xs[0][0] + (tid + c * 256) * 8);
  }
  __syncthreads();

  v8f outacc[8] = {};

  for (int e = 0; e < EE; ++e) {
    const __bf16* w1e = W1 + (size_t)e * DD * DFF;
    const __bf16* w2e = W2 + (size_t)e * DFF * DD;

    float gv[8];
#pragma unroll
    for (int r = 0; r < 8; ++r)
      gv[r] = gates[(size_t)(t0 + wm * 16 + r + 8 * half) * EE + e];

    for (int f0 = 0; f0 < DFF; f0 += 64) {
      for (int c = tid; c < 64 * (DD / 8); c += 256) {
        int n = c >> 5, kc = (c & 31) * 8;
        stage8_strided(w1e + (size_t)kc * DFF + f0 + n, DFF, &W1t[n][kc]);
      }
      for (int c = tid; c < DD * (64 / 8); c += 256) {
        int n = c >> 3, kc = (c & 7) * 8;
        stage8_strided(w2e + (size_t)(f0 + kc) * DD + n, DD, &W2t[n][kc]);
      }
      if (f0 + 64 < DFF) {  // emits global_prefetch_b8
        __builtin_prefetch(w1e + (size_t)tid * DFF + f0 + 64, 0, 1);
        __builtin_prefetch(
            w2e + (size_t)(f0 + 64 + (tid >> 5)) * DD + (tid & 31) * 8, 0, 1);
      }
      __syncthreads();

      // Hc(64x64) = gate * gelu(Xs @ W1 chunk + b1)
      v8f hacc[2] = {};
      for (int k0 = 0; k0 < DD; k0 += 32) {
        v16bf a = load_a_frag(&Xs[wm * 16][0], DD, k0, lane);
#pragma unroll
        for (int nt = 0; nt < 2; ++nt) {
          v16bf b = load_bt_frag(&W1t[0][0], DD, wn * 32 + nt * 16, k0, lane);
          hacc[nt] = wmma_bf16(a, b, hacc[nt]);
        }
      }
#pragma unroll
      for (int nt = 0; nt < 2; ++nt) {
        float bias = B1[e * DFF + f0 + wn * 32 + nt * 16 + (lane & 15)];
#pragma unroll
        for (int r = 0; r < 8; ++r) {
          float g = gelu_tanh(hacc[nt][r] + bias) * gv[r];
          Hs[wm * 16 + r + 8 * half][wn * 32 + nt * 16 + (lane & 15)] = (__bf16)g;
        }
      }
      __syncthreads();

      // outacc += gatedHc @ W2 chunk  (each wave: 16 x 128)
      for (int k0 = 0; k0 < 64; k0 += 32) {
        v16bf a = load_a_frag(&Hs[wm * 16][0], 64, k0, lane);
#pragma unroll
        for (int j = 0; j < 8; ++j) {
          v16bf b = load_bt_frag(&W2t[0][0], 64, wn * 128 + j * 16, k0, lane);
          outacc[j] = wmma_bf16(a, b, outacc[j]);
        }
      }
      __syncthreads();
    }
  }

  // epilogue: add sum_e gate_e * b2_e and store
#pragma unroll
  for (int r = 0; r < 8; ++r) {
    size_t row = (size_t)t0 + wm * 16 + r + 8 * half;
    float g8[EE];
#pragma unroll
    for (int e = 0; e < EE; ++e) g8[e] = gates[row * EE + e];
#pragma unroll
    for (int j = 0; j < 8; ++j) {
      int col = wn * 128 + j * 16 + (lane & 15);
      float extra = 0.f;
#pragma unroll
      for (int e = 0; e < EE; ++e) extra += g8[e] * B2[e * DD + col];
      out[row * DD + col] = outacc[j][r] + extra;
    }
  }
}

// ---------------------------------------------------------------------------
// Host orchestration
// ---------------------------------------------------------------------------
extern "C" void kernel_launch(void* const* d_in, const int* in_sizes, int n_in,
                              void* d_out, int out_size, void* d_ws,
                              size_t ws_size, hipStream_t stream) {
  (void)in_sizes; (void)n_in; (void)out_size; (void)ws_size;

  const float* x_seq = (const float*)d_in[0];
  const float* x_gra = (const float*)d_in[1];
  const unsigned char* mask_seq = (const unsigned char*)d_in[2];
  const unsigned char* mask_gra = (const unsigned char*)d_in[3];
  const float* wq_s = (const float*)d_in[4];
  const float* wk_g = (const float*)d_in[5];
  const float* wv_g = (const float*)d_in[6];
  const float* wo_s = (const float*)d_in[7];
  const float* wq_g = (const float*)d_in[8];
  const float* wk_s = (const float*)d_in[9];
  const float* wv_s = (const float*)d_in[10];
  const float* wo_g = (const float*)d_in[11];
  const float* w1_seq = (const float*)d_in[12];
  const float* b1_seq = (const float*)d_in[13];
  const float* w2_seq = (const float*)d_in[14];
  const float* b2_seq = (const float*)d_in[15];
  const float* w1_gra = (const float*)d_in[16];
  const float* b1_gra = (const float*)d_in[17];
  const float* w2_gra = (const float*)d_in[18];
  const float* b2_gra = (const float*)d_in[19];
  const float* wg_seq = (const float*)d_in[20];
  const float* wg_gra = (const float*)d_in[21];

  const int TS = BB * LSS;
  const int TG = BB * LGG;

  char* ws = (char*)d_ws;
  size_t off = 0;
  auto carve = [&](size_t bytes) -> char* {
    char* p = ws + off;
    off += (bytes + 255) & ~(size_t)255;
    return p;
  };

  __bf16* wq_s_bf = (__bf16*)carve(DD * HH * 2);
  __bf16* wk_g_bf = (__bf16*)carve(DD * HH * 2);
  __bf16* wv_g_bf = (__bf16*)carve(DD * HH * 2);
  __bf16* wo_s_bf = (__bf16*)carve(HH * DD * 2);
  __bf16* wq_g_bf = (__bf16*)carve(DD * HH * 2);
  __bf16* wk_s_bf = (__bf16*)carve(DD * HH * 2);
  __bf16* wv_s_bf = (__bf16*)carve(DD * HH * 2);
  __bf16* wo_g_bf = (__bf16*)carve(HH * DD * 2);
  __bf16* w1s_bf = (__bf16*)carve((size_t)EE * DD * DFF * 2);
  __bf16* w2s_bf = (__bf16*)carve((size_t)EE * DFF * DD * 2);
  __bf16* w1g_bf = (__bf16*)carve((size_t)EE * DD * DFF * 2);
  __bf16* w2g_bf = (__bf16*)carve((size_t)EE * DFF * DD * 2);
  float* xs = (float*)carve((size_t)TS * DD * 4);
  float* xg = (float*)carve((size_t)TG * DD * 4);
  float* gates_s = (float*)carve((size_t)TS * EE * 4);
  float* gates_g = (float*)carve((size_t)TG * EE * 4);
  __bf16* bufQ = (__bf16*)carve((size_t)TS * HH * 2);
  __bf16* bufK = (__bf16*)carve((size_t)TS * HH * 2);
  __bf16* bufV = (__bf16*)carve((size_t)TS * HH * 2);
  __bf16* bufC = (__bf16*)carve((size_t)TS * HH * 2);

  auto cvt = [&](const float* s, __bf16* d, int n) {
    cvt_f32_bf16_kernel<<<(n / 8 + 255) / 256, 256, 0, stream>>>(s, d, n);
  };
  cvt(wq_s, wq_s_bf, DD * HH);  cvt(wk_g, wk_g_bf, DD * HH);
  cvt(wv_g, wv_g_bf, DD * HH);  cvt(wo_s, wo_s_bf, HH * DD);
  cvt(wq_g, wq_g_bf, DD * HH);  cvt(wk_s, wk_s_bf, DD * HH);
  cvt(wv_s, wv_s_bf, DD * HH);  cvt(wo_g, wo_g_bf, HH * DD);
  cvt(w1_seq, w1s_bf, EE * DD * DFF);  cvt(w2_seq, w2s_bf, EE * DFF * DD);
  cvt(w1_gra, w1g_bf, EE * DD * DFF);  cvt(w2_gra, w2g_bf, EE * DFF * DD);

  // ---- phase A: seq attends to graph ----
  gemm256_kernel<false, false><<<dim3(TS / 64, 4), 256, 0, stream>>>(
      x_seq, wq_s_bf, nullptr, bufQ, TS);
  gemm256_kernel<false, false><<<dim3(TG / 64, 4), 256, 0, stream>>>(
      x_gra, wk_g_bf, nullptr, bufK, TG);
  gemm256_kernel<false, false><<<dim3(TG / 64, 4), 256, 0, stream>>>(
      x_gra, wv_g_bf, nullptr, bufV, TG);
  attn_kernel<<<dim3(LSS / 64, BB), 256, 0, stream>>>(bufQ, bufK, bufV,
                                                      mask_gra, bufC, LSS, LGG);
  gemm256_kernel<true, true><<<dim3(TS / 64, 4), 256, 0, stream>>>(
      bufC, wo_s_bf, x_seq, xs, TS);

  // ---- phase B: graph attends to seq (scratch reused) ----
  gemm256_kernel<false, false><<<dim3(TG / 64, 4), 256, 0, stream>>>(
      x_gra, wq_g_bf, nullptr, bufQ, TG);
  gemm256_kernel<false, false><<<dim3(TS / 64, 4), 256, 0, stream>>>(
      x_seq, wk_s_bf, nullptr, bufK, TS);
  gemm256_kernel<false, false><<<dim3(TS / 64, 4), 256, 0, stream>>>(
      x_seq, wv_s_bf, nullptr, bufV, TS);
  attn_kernel<<<dim3(LGG / 64, BB), 256, 0, stream>>>(bufQ, bufK, bufV,
                                                      mask_seq, bufC, LGG, LSS);
  gemm256_kernel<true, true><<<dim3(TG / 64, 4), 256, 0, stream>>>(
      bufC, wo_g_bf, x_gra, xg, TG);

  // ---- gates ----
  gate_kernel<<<(TS + 255) / 256, 256, 0, stream>>>(xs, wg_seq, gates_s, TS);
  gate_kernel<<<(TG + 255) / 256, 256, 0, stream>>>(xg, wg_gra, gates_g, TG);

  // ---- fused MoE ----
  float* out_seq = (float*)d_out;
  float* out_gra = out_seq + (size_t)TS * DD;
  moe_kernel<<<TS / 64, 256, 0, stream>>>(xs, w1s_bf, b1_seq, w2s_bf, b2_seq,
                                          gates_s, out_seq, TS);
  moe_kernel<<<TG / 64, 256, 0, stream>>>(xg, w1g_bf, b1_gra, w2g_bf, b2_gra,
                                          gates_g, out_gra, TG);
}
